// EGNN_19662360281098
// MI455X (gfx1250) — matrix operand order repaired and checked by reading, not compiled, not run on hardware
//
#include <hip/hip_runtime.h>
#include <hip/hip_bf16.h>

typedef __attribute__((ext_vector_type(16))) _Float16 v16h;
typedef __attribute__((ext_vector_type(2)))  _Float16 h2;
typedef __attribute__((ext_vector_type(8)))  float    v8f;

#define NNODE 1024
#define DIMF  64
#define HPAD  288   // hidden dim 258 padded to 9*32
#define MDIM  16

union AU { v16h v; h2 p[8]; };        // A fragment as 8 f16-pairs
union BU { v16h v; uint4 u4[2]; };    // B fragment as 2x b128
union RU { uint4 u; h2 p[4]; };       // one 16B run = 8 halves

// Same-wave LDS RAW fence: LDS ops from one wave complete in order (DScnt),
// so s_wait_dscnt 0 is sufficient for wave-private LDS regions (no barrier).
__device__ __forceinline__ void wave_lds_fence() {
    asm volatile("s_wait_dscnt 0x0" ::: "memory");
}

// ---------------------------------------------------------------------------
// Prep 1: AfH[i,c] = f16(feats[i]@eW1[0:64,c] + eb1[c])
//         BfH[i,c] = f16(feats[i]@eW1[64:128,c])
//         wlH[c]   = f16(eW1[128,c]); zero-padded to HPAD channels.
// ---------------------------------------------------------------------------
__global__ void egnn_prep_ab(const float* __restrict__ feats,
                             const float* __restrict__ eW1,
                             const float* __restrict__ eb1,
                             _Float16* __restrict__ AfH, _Float16* __restrict__ BfH,
                             _Float16* __restrict__ wlH) {
    const int i = blockIdx.x;        // node
    const int c = threadIdx.x;       // 0..287
    const int which = blockIdx.y;    // 0 -> Af, 1 -> Bf
    float acc = 0.f;
    if (c < 258) {
        #pragma unroll 4
        for (int k = 0; k < 64; ++k)
            acc += feats[i * 64 + k] * eW1[(which * 64 + k) * 258 + c];
        if (which == 0) acc += eb1[c];
    }
    (which == 0 ? AfH : BfH)[i * HPAD + c] = (_Float16)acc;
    if (i == 0 && which == 0)
        wlH[c] = (c < 258) ? (_Float16)eW1[128 * 258 + c] : (_Float16)0.f;
}

// ---------------------------------------------------------------------------
// Prep 2: pack eW2 (258x16 -> 288x16 padded) and cW1 (16x64, K padded to 32)
// into the WMMA 16-bit B-matrix lane layout:
//   lane l: N = l%16,  K = (v/4)*16 + (l/16)*8 + 2(v%4) + {0,1}   (v = VGPR index 0..7)
// eW2p: [9 chunks][32 lanes][8 dwords], cW1p: [4 tiles][32 lanes][8 dwords]
// ---------------------------------------------------------------------------
__global__ void egnn_prep_pack(const float* __restrict__ eW2,
                               const float* __restrict__ cW1,
                               unsigned* __restrict__ eW2p,
                               unsigned* __restrict__ cW1p) {
    const int t = blockIdx.x * blockDim.x + threadIdx.x;
    union { unsigned u; _Float16 f[2]; } pk;

    if (t < 9 * 32 * 8) {            // eW2 pack
        const int v = t & 7, lane = (t >> 3) & 31, q = t >> 8;
        const int n = lane & 15;
        const int k = q * 32 + (v >> 2) * 16 + (lane >> 4) * 8 + 2 * (v & 3);
        pk.f[0] = (k     < 258) ? (_Float16)eW2[(k)     * 16 + n] : (_Float16)0.f;
        pk.f[1] = (k + 1 < 258) ? (_Float16)eW2[(k + 1) * 16 + n] : (_Float16)0.f;
        eW2p[t] = pk.u;
    }
    if (t < 4 * 32 * 8) {            // cW1 pack (K=16 real, padded to 32)
        const int v = t & 7, lane = (t >> 3) & 31, tt = t >> 8;
        const int n = tt * 16 + (lane & 15);
        const int k = (v >> 2) * 16 + (lane >> 4) * 8 + 2 * (v & 3);
        pk.f[0] = (k     < 16) ? (_Float16)cW1[(k)     * 64 + n] : (_Float16)0.f;
        pk.f[1] = (k + 1 < 16) ? (_Float16)cW1[(k + 1) * 64 + n] : (_Float16)0.f;
        cW1p[t] = pk.u;
    }
}

// ---------------------------------------------------------------------------
// Main fused kernel: one block per node i (256 threads = 8 waves).
// Each wave handles strips of 16 edges (i, j0..j0+15), 8 strips per wave,
// fully independently (wave-private LDS, no per-strip workgroup barriers).
// A-fragment channels per lane form two contiguous 8-channel runs per chunk:
//   run0 = q*32 + 8g .. +7  (VGPRs 0..3),  run1 = q*32 + 16 + 8g .. +7 (VGPRs 4..7)
// so each operand needs just 2x b128 loads per chunk in f16.
// ---------------------------------------------------------------------------
__launch_bounds__(256)
__global__ void egnn_edge_kernel(
    const float* __restrict__ feats, const float* __restrict__ coors,
    const _Float16* __restrict__ AfH, const _Float16* __restrict__ BfH,
    const _Float16* __restrict__ wlH,
    const unsigned* __restrict__ eW2p, const unsigned* __restrict__ cW1p,
    const float* __restrict__ eb2, const float* __restrict__ cb1,
    const float* __restrict__ cW2, const float* __restrict__ cb2,
    const float* __restrict__ hW1, const float* __restrict__ hb1,
    const float* __restrict__ hW2, const float* __restrict__ hb2,
    float* __restrict__ out_h, float* __restrict__ out_c) {

    __shared__ float lds_m[8][16][16];   // per-wave m_ij staging (edge x channel)
    __shared__ float lds_w[8][16];       // per-wave coor_weights
    __shared__ float lds_mi[8][16];      // per-wave m_i partials
    __shared__ float lds_c[8][3];        // per-wave coors partials
    __shared__ float lds_x[80];          // [feats_i | m_i]
    __shared__ float lds_hid[128];       // hidden_mlp layer1

    const int i    = blockIdx.x;
    const int tid  = threadIdx.x;
    const int wave = tid >> 5;
    const int lane = tid & 31;
    const int n    = lane & 15;          // channel / row-within-tile index
    const int g    = lane >> 4;          // lane half

    const float cix = coors[i * 3 + 0];
    const float ciy = coors[i * 3 + 1];
    const float ciz = coors[i * 3 + 2];
    const float eb2n = eb2[n];
    const float cb2s = cb2[0];
    const _Float16 hzero = (_Float16)0.f;

    float macc = 0.f;                    // m_i accumulator for channel n
    float cax = 0.f, cay = 0.f, caz = 0.f;

    for (int it = 0; it < 8; ++it) {
        const int j0 = it * 128 + wave * 16;
        const int jm = j0 + n;           // this lane's A-matrix row (edge j)

        const float rx = cix - coors[jm * 3 + 0];
        const float ry = ciy - coors[jm * 3 + 1];
        const float rz = ciz - coors[jm * 3 + 2];
        const float dist = sqrtf(rx * rx + ry * ry + rz * rz);
        h2 d2; d2.x = (_Float16)dist; d2.y = (_Float16)dist;

        // ---- m_ij = relu(hidden1) @ eW2 + eb2, via 9 chained WMMAs --------
        v8f c = {0, 0, 0, 0, 0, 0, 0, 0};
        #pragma unroll 3
        for (int q = 0; q < 9; ++q) {
            const int r0 = q * 32 + 8 * g;        // run0 channel base
            const int r1 = r0 + 16;               // run1 channel base
            RU af0, af1, bf0, bf1, w0, w1;
            af0.u = *(const uint4*)(AfH + i  * HPAD + r0);
            af1.u = *(const uint4*)(AfH + i  * HPAD + r1);
            bf0.u = *(const uint4*)(BfH + jm * HPAD + r0);
            bf1.u = *(const uint4*)(BfH + jm * HPAD + r1);
            w0.u  = *(const uint4*)(wlH + r0);
            w1.u  = *(const uint4*)(wlH + r1);

            AU a;
            #pragma unroll
            for (int p = 0; p < 4; ++p) {
                h2 t0 = af0.p[p] + bf0.p[p] + d2 * w0.p[p];
                h2 t1 = af1.p[p] + bf1.p[p] + d2 * w1.p[p];
                t0.x = t0.x > hzero ? t0.x : hzero;
                t0.y = t0.y > hzero ? t0.y : hzero;
                t1.x = t1.x > hzero ? t1.x : hzero;
                t1.y = t1.y > hzero ? t1.y : hzero;
                a.p[p]     = t0;
                a.p[4 + p] = t1;
            }

            BU b;
            const uint4* bp = (const uint4*)(eW2p + (q * 32 + lane) * 8);
            b.u4[0] = bp[0];
            b.u4[1] = bp[1];

            c = __builtin_amdgcn_wmma_f32_16x16x32_f16(false, a.v, false, b.v,
                                                       (short)0, c, false, false);
        }
        #pragma unroll
        for (int r = 0; r < 8; ++r) c[r] += eb2n;

        // ---- m_i partial: sum m_ij over the 16 edges ----------------------
        {
            float s = 0.f;
            #pragma unroll
            for (int r = 0; r < 8; ++r) s += c[r];
            s += __shfl_xor(s, 16, 32);
            macc += s;
        }

        // ---- stage m_ij to wave-private LDS as [edge][channel] ------------
        #pragma unroll
        for (int r = 0; r < 8; ++r) lds_m[wave][r + 8 * g][n] = c[r];
        wave_lds_fence();   // same-wave RAW: LDS is in-order, DScnt==0 suffices

        // ---- coors MLP: relu(m_ij @ cW1 + cb1) @ cW2 + cb2 ----------------
        AU am;
        #pragma unroll
        for (int v = 0; v < 8; ++v) {
            h2 t;
            if (v < 4) {
                const int k = 2 * v + 8 * g;
                t.x = (_Float16)lds_m[wave][n][k];
                t.y = (_Float16)lds_m[wave][n][k + 1];
            } else {
                t.x = hzero; t.y = hzero;
            }
            am.p[v] = t;
        }
        float wsum[8];
        #pragma unroll
        for (int r = 0; r < 8; ++r) wsum[r] = 0.f;
        #pragma unroll
        for (int t = 0; t < 4; ++t) {
            BU bt;
            const uint4* bp = (const uint4*)(cW1p + (t * 32 + lane) * 8);
            bt.u4[0] = bp[0];
            bt.u4[1] = bp[1];
            v8f z = {0, 0, 0, 0, 0, 0, 0, 0};
            v8f d = __builtin_amdgcn_wmma_f32_16x16x32_f16(false, am.v, false, bt.v,
                                                           (short)0, z, false, false);
            const float cb = cb1[t * 16 + n];
            const float cw = cW2[t * 16 + n];
            #pragma unroll
            for (int r = 0; r < 8; ++r)
                wsum[r] += fmaxf(d[r] + cb, 0.f) * cw;
        }
        // butterfly reduce over the 64 hidden channels (16 lanes per half)
        #pragma unroll
        for (int r = 0; r < 8; ++r) {
            wsum[r] += __shfl_xor(wsum[r], 1, 32);
            wsum[r] += __shfl_xor(wsum[r], 2, 32);
            wsum[r] += __shfl_xor(wsum[r], 4, 32);
            wsum[r] += __shfl_xor(wsum[r], 8, 32);
        }
        if (n == 0) {
            #pragma unroll
            for (int r = 0; r < 8; ++r)
                lds_w[wave][r + 8 * g] = wsum[r] + cb2s;
        }
        wave_lds_fence();   // same-wave cross-lane RAW through private LDS

        // ---- coors accumulation: lanes 0..15 own edge (i, j0+lane) --------
        if (lane < 16) {
            const float w = lds_w[wave][lane];
            cax += w * rx; cay += w * ry; caz += w * rz;
        }
    }

    // ---- block-level reductions (genuine cross-wave: barrier needed) -----
    if (lane < 16) lds_mi[wave][lane] = macc;
    cax += __shfl_xor(cax, 1, 32); cay += __shfl_xor(cay, 1, 32); caz += __shfl_xor(caz, 1, 32);
    cax += __shfl_xor(cax, 2, 32); cay += __shfl_xor(cay, 2, 32); caz += __shfl_xor(caz, 2, 32);
    cax += __shfl_xor(cax, 4, 32); cay += __shfl_xor(cay, 4, 32); caz += __shfl_xor(caz, 4, 32);
    cax += __shfl_xor(cax, 8, 32); cay += __shfl_xor(cay, 8, 32); caz += __shfl_xor(caz, 8, 32);
    if (lane == 0) { lds_c[wave][0] = cax; lds_c[wave][1] = cay; lds_c[wave][2] = caz; }
    __syncthreads();

    if (tid < 64) {
        lds_x[tid] = feats[i * 64 + tid];
    } else if (tid < 80) {
        const int ch = tid - 64;
        float s = 0.f;
        #pragma unroll
        for (int w = 0; w < 8; ++w) s += lds_mi[w][ch];
        lds_x[64 + ch] = s;
    }
    if (tid < 3) {
        float s = 0.f;
        #pragma unroll
        for (int w = 0; w < 8; ++w) s += lds_c[w][tid];
        out_c[i * 3 + tid] = s;
    }
    __syncthreads();

    // ---- hidden MLP: relu([feats_i|m_i] @ hW1 + hb1) @ hW2 + hb2 ---------
    if (tid < 128) {
        float h = hb1[tid];
        #pragma unroll 4
        for (int k = 0; k < 80; ++k) h += lds_x[k] * hW1[k * 128 + tid];
        lds_hid[tid] = fmaxf(h, 0.f);
    }
    __syncthreads();
    if (tid < 64) {
        float o = hb2[tid];
        #pragma unroll 4
        for (int k = 0; k < 128; ++k) o += lds_hid[k] * hW2[k * 64 + tid];
        out_h[i * 64 + tid] = o;
    }
}

// ---------------------------------------------------------------------------
extern "C" void kernel_launch(void* const* d_in, const int* in_sizes, int n_in,
                              void* d_out, int out_size, void* d_ws, size_t ws_size,
                              hipStream_t stream) {
    (void)in_sizes; (void)n_in; (void)out_size; (void)ws_size;
    const float* feats = (const float*)d_in[0];
    const float* coors = (const float*)d_in[1];
    const float* eW1   = (const float*)d_in[2];
    const float* eb1   = (const float*)d_in[3];
    const float* eW2   = (const float*)d_in[4];
    const float* eb2   = (const float*)d_in[5];
    const float* cW1   = (const float*)d_in[6];
    const float* cb1   = (const float*)d_in[7];
    const float* cW2   = (const float*)d_in[8];
    const float* cb2   = (const float*)d_in[9];
    const float* hW1   = (const float*)d_in[10];
    const float* hb1   = (const float*)d_in[11];
    const float* hW2   = (const float*)d_in[12];
    const float* hb2   = (const float*)d_in[13];

    float* out   = (float*)d_out;
    float* out_h = out;                  // 1024*64 floats
    float* out_c = out + NNODE * DIMF;   // 1024*3 floats

    _Float16* AfH = (_Float16*)d_ws;             // 1024*288 halves
    _Float16* BfH = AfH + NNODE * HPAD;          // 1024*288 halves
    _Float16* wlH = BfH + NNODE * HPAD;          // 288 halves
    unsigned* eW2p = (unsigned*)(wlH + HPAD);    // 9*32*8 dwords (16B aligned)
    unsigned* cW1p = eW2p + 9 * 32 * 8;          // 4*32*8 dwords

    egnn_prep_ab<<<dim3(NNODE, 2), HPAD, 0, stream>>>(feats, eW1, eb1, AfH, BfH, wlH);
    egnn_prep_pack<<<(9 * 32 * 8 + 255) / 256, 256, 0, stream>>>(eW2, cW1, eW2p, cW1p);
    egnn_edge_kernel<<<NNODE, 256, 0, stream>>>(feats, coors, AfH, BfH, wlH, eW2p, cW1p,
                                                eb2, cb1, cW2, cb2,
                                                hW1, hb1, hW2, hb2, out_h, out_c);
}
